// EMO_25967372271808
// MI455X (gfx1250) — compile-verified, matrix operand order checked by reading, not run on hardware
//
#include <hip/hip_runtime.h>
#include <hip/hip_bf16.h>

// ---------------- types ----------------
typedef __attribute__((ext_vector_type(16))) _Float16 v16h;
typedef __attribute__((ext_vector_type(8)))  float    v8f;

#define CC   256
#define BB   32
#define HH   56
#define WW2  56
#define HWIMG 3136       // 56*56
#define NWIN 2048        // 32*8*8
#define NPIX 49          // 7*7 window pixels
#define NPAD 64

// LDS leading dims (halves unless noted)
#define LDXP 264         // x window, pixel-major [64][LDXP]
#define LDQK 40          // q/k per head [64][LDQK]
#define LDVB 72          // v (B-layout) per head [32][LDVB]
#define LDP  72          // P per head [16][LDP]
#define LDO  33          // O per head [49][LDO] floats

// LDS byte offsets inside attn kernel
#define OFF_XP 0
#define OFF_Q  (OFF_XP + 64*LDXP*2)               // 33792
#define OFF_K  (OFF_Q  + 8*64*LDQK*2)             // 74752
#define OFF_V  (OFF_K  + 8*64*LDQK*2)             // 115712
#define OFF_P  (OFF_V  + 8*32*LDVB*2)             // 152576
#define OFF_O  (OFF_P  + 8*16*LDP*2)              // 171008
#define OFF_WC (OFF_O  + 8*49*LDO*4)              // 222752
#define SMEM_ATTN (OFF_WC + 256*84*4)             // 308768 bytes

__device__ __forceinline__ v8f vzero8() {
  v8f z;
#pragma unroll
  for (int i = 0; i < 8; ++i) z[i] = 0.0f;
  return z;
}

// CDNA5 async global->LDS copy (lane-addressed, tracked by ASYNCcnt).
// VDST = wave-relative LDS byte offset (low 32 bits of generic shared ptr),
// VADDR = 64-bit global address, no SADDR.
__device__ __forceinline__ void async_g2l_b32(void* lds, const void* g) {
  unsigned l = (unsigned)(unsigned long long)lds;
  asm volatile("global_load_async_to_lds_b32 %0, %1, off"
               :: "v"(l), "v"(g) : "memory");
}
__device__ __forceinline__ void wait_async0() {
  asm volatile("s_wait_asynccnt 0" ::: "memory");
}

// Load a 16x32 f16 fragment.
// A-operand: matrix stored row-major [m][K], lane holds row m = lane&15.
// B-operand: store B^T row-major [n][K]; same striping (symmetric layout).
// ISA 16-bit A 16x32 layout: lanes 0-15 hold K {0..7,16..23}, lanes 16-31 hold
// K {8..15,24..31}; VGPR j holds k pair (j&3)*2 + (j>>2)*16 (+8 for hi half).
__device__ __forceinline__ v16h load_frag(const _Float16* __restrict__ base,
                                          int ld, int kbase, int lane) {
  const int half = lane >> 4;
  const int r    = lane & 15;
  const _Float16* p = base + r * ld + kbase + half * 8;
  v16h out;
#pragma unroll
  for (int i = 0; i < 8; ++i) {
    out[i]     = p[i];        // K = h*8 + i
    out[8 + i] = p[16 + i];   // K = 16 + h*8 + i
  }
  return out;
}

__device__ __forceinline__ v8f wmma_f16(v16h a, v16h b, v8f c) {
  return __builtin_amdgcn_wmma_f32_16x16x32_f16(false, a, false, b,
                                                (short)0, c, false, false);
}

// ---------------- K1: BN batch statistics ----------------
__global__ void bn_stats(const float* __restrict__ x,
                         const float* __restrict__ gamma,
                         const float* __restrict__ beta,
                         float* __restrict__ scale,
                         float* __restrict__ shift) {
  __shared__ float ss[256], sq[256];
  const int c = blockIdx.x, tid = threadIdx.x;
  float s = 0.f, q = 0.f;
  for (int b = 0; b < BB; ++b) {
    const float* p = x + ((size_t)(b * CC + c)) * HWIMG;
    for (int i = tid; i < HWIMG; i += 256) {
      float v = p[i];
      s += v; q += v * v;
    }
  }
  ss[tid] = s; sq[tid] = q;
  __syncthreads();
  for (int st = 128; st > 0; st >>= 1) {
    if (tid < st) { ss[tid] += ss[tid + st]; sq[tid] += sq[tid + st]; }
    __syncthreads();
  }
  if (tid == 0) {
    const float N = (float)(BB * HWIMG);
    float mean = ss[0] / N;
    float var  = sq[0] / N - mean * mean;
    float inv  = rsqrtf(var + 1e-5f);
    float sc   = gamma[c] * inv;
    scale[c] = sc;
    shift[c] = beta[c] - mean * sc;
  }
}

// ---------------- K2: pack weights to f16 ----------------
__global__ void pack_w(const float* __restrict__ qkv_w,
                       const float* __restrict__ w_in,
                       const float* __restrict__ w_out,
                       _Float16* __restrict__ wq,
                       _Float16* __restrict__ wi,
                       _Float16* __restrict__ wo) {
  int i = blockIdx.x * 256 + threadIdx.x;
  if (i < 768 * 256)            wq[i]               = (_Float16)qkv_w[i];
  else if (i < 768*256 + 65536) wi[i - 768*256]     = (_Float16)w_in[i - 768*256];
  else if (i < 768*256 + 131072)wo[i - 768*256 - 65536] = (_Float16)w_out[i - 768*256 - 65536];
}

// ---------------- K3: fused window attention ----------------
__global__ void attn_win(const float* __restrict__ x,
                         const float* __restrict__ scale,
                         const float* __restrict__ shift,
                         const _Float16* __restrict__ wqkv,
                         const float* __restrict__ qkv_b,
                         const float* __restrict__ w3, const float* __restrict__ b3,
                         const float* __restrict__ w5, const float* __restrict__ b5,
                         const float* __restrict__ w7, const float* __restrict__ b7,
                         float* __restrict__ xa) {
  extern __shared__ char smem[];
  _Float16* s_xp = (_Float16*)(smem + OFF_XP);   // [64][LDXP] (p-major, k=c)
  _Float16* s_q  = (_Float16*)(smem + OFF_Q);    // 8 x [64][LDQK]
  _Float16* s_k  = (_Float16*)(smem + OFF_K);    // 8 x [64][LDQK]
  _Float16* s_v  = (_Float16*)(smem + OFF_V);    // 8 x [32][LDVB] (n=d, k=p')
  _Float16* s_P  = (_Float16*)(smem + OFF_P);    // 8 x [16][LDP]
  float*    s_O  = (float*)   (smem + OFF_O);    // 8 x [49][LDO]
  float*    s_wc = (float*)   (smem + OFF_WC);   // [256][84] dw weights

  const int tid  = threadIdx.x;
  const int lane = tid & 31;
  const int wave = tid >> 5;
  const int half = lane >> 4;
  const int nl   = lane & 15;

  const int win = blockIdx.x;
  const int b  = win >> 6;
  const int i1 = (win >> 3) & 7;
  const int i2 = win & 7;

  // stage dw-conv weights (3x3 | 5x5 | 7x7 concatenated per channel) using
  // CDNA5 async global->LDS copies (ASYNCcnt path, no VGPR round-trip).
  for (int i = tid; i < 256 * 83; i += 256) {
    int c = i / 83, t = i % 83;
    const float* src;
    if (t < 9)       src = w3 + c * 9 + t;
    else if (t < 34) src = w5 + c * 25 + (t - 9);
    else             src = w7 + c * 49 + (t - 34);
    async_g2l_b32(s_wc + c * 84 + t, src);
  }

  // load + BN the window: thread == channel
  {
    const int c = tid;
    const float sc = scale[c], sh = shift[c];
    const float* xb = x + ((size_t)(b * CC + c)) * HWIMG;
#pragma unroll
    for (int p = 0; p < NPIX; ++p) {
      int hh = (p / 7) * 8 + i1;
      int ww = (p % 7) * 8 + i2;
      s_xp[p * LDXP + c] = (_Float16)(xb[hh * WW2 + ww] * sc + sh);
    }
    for (int p = NPIX; p < NPAD; ++p) s_xp[p * LDXP + c] = (_Float16)0.f;
  }
  wait_async0();          // async LDS writes are NOT ordered by the barrier
  __syncthreads();

  // ---- QKV GEMM: out[768][64] = Wqkv[768][256] * xw[256][64] ----
  // 48 M-tiles x 4 N-tiles = 192 tiles, 24 per wave, K=256 in 8 steps.
  for (int t = wave * 24; t < wave * 24 + 24; ++t) {
    const int mt = t >> 2, nt = t & 3;
    v8f acc = vzero8();
#pragma unroll
    for (int kk = 0; kk < 256; kk += 32) {
      v16h a  = load_frag(wqkv + (size_t)(mt * 16) * 256, 256, kk, lane);
      v16h bf = load_frag(s_xp + nt * 16 * LDXP, LDXP, kk, lane);
      acc = wmma_f16(a, bf, acc);
    }
#pragma unroll
    for (int j = 0; j < 8; ++j) {
      const int o = mt * 16 + j + 8 * half;
      const int p = nt * 16 + nl;
      float val = acc[j] + qkv_b[o];
      const int c = o & 255, head = c >> 5, d = c & 31;
      if (o < 256) {
        s_q[head * 64 * LDQK + p * LDQK + d] = (_Float16)val;
      } else if (o < 512) {
        s_k[head * 64 * LDQK + p * LDQK + d] = (_Float16)val;
      } else {
        val += (float)s_xp[p * LDXP + c];       // v-residual
        s_v[head * 32 * LDVB + d * LDVB + p] = (_Float16)val;
      }
    }
  }
  __syncthreads();

  // ---- per-head attention: wave w handles head w ----
  {
    const int head = wave;
    _Float16* qh = s_q + head * 64 * LDQK;
    _Float16* kh = s_k + head * 64 * LDQK;
    _Float16* vh = s_v + head * 32 * LDVB;
    _Float16* Ph = s_P + head * 16 * LDP;
    float*    Oh = s_O + head * 49 * LDO;
    const float sc = 0.17677669529663687f;      // 1/sqrt(32)

    for (int mb = 0; mb < 4; ++mb) {
      v8f sacc[4];
      v16h a = load_frag(qh + mb * 16 * LDQK, LDQK, 0, lane);
#pragma unroll
      for (int nt = 0; nt < 4; ++nt) {
        sacc[nt] = vzero8();
        v16h bf = load_frag(kh + nt * 16 * LDQK, LDQK, 0, lane);
        sacc[nt] = wmma_f16(a, bf, sacc[nt]);
      }
      float rsum[8];
#pragma unroll
      for (int j = 0; j < 8; ++j) {
        float mx = -3.0e38f;
#pragma unroll
        for (int nt = 0; nt < 4; ++nt) {
          int n = nt * 16 + nl;
          float s = (n < NPIX) ? sacc[nt][j] * sc : -3.0e38f;
          sacc[nt][j] = s;
          mx = fmaxf(mx, s);
        }
        mx = fmaxf(mx, __shfl_xor(mx, 1, 32));
        mx = fmaxf(mx, __shfl_xor(mx, 2, 32));
        mx = fmaxf(mx, __shfl_xor(mx, 4, 32));
        mx = fmaxf(mx, __shfl_xor(mx, 8, 32));
        float su = 0.f;
#pragma unroll
        for (int nt = 0; nt < 4; ++nt) {
          int n = nt * 16 + nl;
          float e = (n < NPIX) ? __expf(sacc[nt][j] - mx) : 0.f;
          sacc[nt][j] = e;
          su += e;
        }
        su += __shfl_xor(su, 1, 32);
        su += __shfl_xor(su, 2, 32);
        su += __shfl_xor(su, 4, 32);
        su += __shfl_xor(su, 8, 32);
        rsum[j] = su;
      }
      // store unnormalized P (f16) for the PV GEMM
#pragma unroll
      for (int j = 0; j < 8; ++j)
#pragma unroll
        for (int nt = 0; nt < 4; ++nt)
          Ph[(j + 8 * half) * LDP + nt * 16 + nl] = (_Float16)sacc[nt][j];

      // O_block[16][32] = P[16][64] * V[64][32]
      v8f o0 = vzero8(), o1 = vzero8();
#pragma unroll
      for (int ks = 0; ks < 2; ++ks) {
        v16h pa = load_frag(Ph, LDP, ks * 32, lane);
        v16h b0 = load_frag(vh, LDVB, ks * 32, lane);
        v16h b1 = load_frag(vh + 16 * LDVB, LDVB, ks * 32, lane);
        o0 = wmma_f16(pa, b0, o0);
        o1 = wmma_f16(pa, b1, o1);
      }
#pragma unroll
      for (int j = 0; j < 8; ++j) {
        const int m = mb * 16 + j + 8 * half;
        if (m < NPIX) {
          float r = 1.0f / rsum[j];
          Oh[m * LDO + nl]      = o0[j] * r;
          Oh[m * LDO + 16 + nl] = o1[j] * r;
        }
      }
    }
  }

  // ---- combine: 0.25*(attn + dw3 + dw5 + dw7) + v; un-window store ----
  {
    const int c = tid, head = c >> 5, d = c & 31;
    const float*    Oh  = s_O + head * 49 * LDO;
    const _Float16* vh  = s_v + head * 32 * LDVB;
    const float*    wcp = s_wc + c * 84;
    const float cb = b3[c] + b5[c] + b7[c];
    float* xab = xa + ((size_t)(b * CC + c)) * HWIMG;
    for (int p = 0; p < NPIX; ++p) {
      const int ph = p / 7, pw = p % 7;
      float s3 = 0.f, s5 = 0.f, s7 = 0.f;
#pragma unroll
      for (int kh = -3; kh <= 3; ++kh) {
        const int yy = ph + kh;
        if ((unsigned)yy >= 7u) continue;
#pragma unroll
        for (int kw = -3; kw <= 3; ++kw) {
          const int xx = pw + kw;
          if ((unsigned)xx >= 7u) continue;
          const float xv = (float)s_xp[(yy * 7 + xx) * LDXP + c];
          s7 += xv * wcp[34 + (kh + 3) * 7 + (kw + 3)];
          if (kh >= -2 && kh <= 2 && kw >= -2 && kw <= 2)
            s5 += xv * wcp[9 + (kh + 2) * 5 + (kw + 2)];
          if (kh >= -1 && kh <= 1 && kw >= -1 && kw <= 1)
            s3 += xv * wcp[(kh + 1) * 3 + (kw + 1)];
        }
      }
      const float vres = (float)vh[d * LDVB + p];
      const float out  = 0.25f * (Oh[p * LDO + d] + s3 + s5 + s7 + cb) + vres;
      xab[(ph * 8 + i1) * WW2 + (pw * 8 + i2)] = out;
    }
  }
}

// ---------------- K4/K6: FFN 1x1-conv GEMM (M=256, Ntile=128, K=256) ----------------
template <typename TI>
__global__ void ffn_gemm(const TI* __restrict__ in,
                         const _Float16* __restrict__ wgt,   // [256][256] f16
                         const float* __restrict__ bias,
                         const float* __restrict__ resid,    // nullable
                         float* __restrict__ out_f32,        // nullable
                         _Float16* __restrict__ out_f16) {   // nullable
  __shared__ _Float16 sB[128 * 40];
  const int tid  = threadIdx.x;
  const int lane = tid & 31;
  const int wave = tid >> 5;
  const int half = lane >> 4;
  const int nl   = lane & 15;
  const size_t nbase = (size_t)blockIdx.x * 128;

  v8f acc[16];
#pragma unroll
  for (int i = 0; i < 16; ++i) acc[i] = vzero8();

  for (int kk = 0; kk < 256; kk += 32) {
    __syncthreads();
#pragma unroll
    for (int i = 0; i < 16; ++i) {
      const int idx = i * 256 + tid;           // 0..4095
      const int n  = idx & 127;
      const int cl = idx >> 7;                 // 0..31
      const size_t g = nbase + n;
      const int bb = (int)(g / HWIMG);
      const int hw = (int)(g % HWIMG);
      const float v = (float)in[((size_t)bb * CC + kk + cl) * HWIMG + hw];
      sB[n * 40 + cl] = (_Float16)v;
    }
    __syncthreads();
    if (kk + 32 < 256)
      __builtin_prefetch(wgt + (size_t)(wave * 32) * 256 + kk + 32, 0, 1);
#pragma unroll
    for (int mt = 0; mt < 2; ++mt) {
      const int mrow = wave * 32 + mt * 16;
      v16h a = load_frag(wgt + (size_t)mrow * 256, 256, kk, lane);
#pragma unroll
      for (int nt = 0; nt < 8; ++nt) {
        v16h bf = load_frag(sB + nt * 16 * 40, 40, 0, lane);
        acc[mt * 8 + nt] = wmma_f16(a, bf, acc[mt * 8 + nt]);
      }
    }
  }
  // epilogue
#pragma unroll
  for (int mt = 0; mt < 2; ++mt)
#pragma unroll
    for (int nt = 0; nt < 8; ++nt)
#pragma unroll
      for (int j = 0; j < 8; ++j) {
        const int o = wave * 32 + mt * 16 + j + 8 * half;
        const size_t g = nbase + nt * 16 + nl;
        const int bb = (int)(g / HWIMG);
        const int hw = (int)(g % HWIMG);
        const size_t addr = ((size_t)bb * CC + o) * HWIMG + hw;
        float v = acc[mt * 8 + nt][j] + bias[o];
        if (resid)   v += resid[addr];
        if (out_f32) out_f32[addr] = v;
        else         out_f16[addr] = (_Float16)v;
      }
}

// ---------------- K5: image-space dw3x3 + SiLU ----------------
__global__ void dw_silu(const _Float16* __restrict__ y1,
                        const float* __restrict__ w,     // (C,1,3,3)
                        const float* __restrict__ bias,
                        _Float16* __restrict__ y2) {
  const size_t i = (size_t)blockIdx.x * 256 + threadIdx.x;
  const int hw = (int)(i % HWIMG);
  const int bc = (int)(i / HWIMG);
  const int c  = bc & 255;
  const int yy = hw / WW2, xx = hw % WW2;
  float acc = bias[c];
#pragma unroll
  for (int kh = -1; kh <= 1; ++kh) {
    const int y = yy + kh;
    if ((unsigned)y >= (unsigned)HH) continue;
#pragma unroll
    for (int kw = -1; kw <= 1; ++kw) {
      const int x = xx + kw;
      if ((unsigned)x >= (unsigned)WW2) continue;
      acc += (float)y1[(size_t)bc * HWIMG + y * WW2 + x] *
             w[c * 9 + (kh + 1) * 3 + (kw + 1)];
    }
  }
  const float s = acc / (1.0f + __expf(-acc));   // SiLU
  y2[i] = (_Float16)s;
}

// ---------------- launcher ----------------
extern "C" void kernel_launch(void* const* d_in, const int* in_sizes, int n_in,
                              void* d_out, int out_size, void* d_ws, size_t ws_size,
                              hipStream_t stream) {
  const float* x        = (const float*)d_in[0];
  const float* bn_gamma = (const float*)d_in[1];
  const float* bn_beta  = (const float*)d_in[2];
  const float* qkv_w    = (const float*)d_in[3];
  const float* qkv_b    = (const float*)d_in[4];
  const float* conv3_w  = (const float*)d_in[5];
  const float* conv3_b  = (const float*)d_in[6];
  const float* conv5_w  = (const float*)d_in[7];
  const float* conv5_b  = (const float*)d_in[8];
  const float* conv7_w  = (const float*)d_in[9];
  const float* conv7_b  = (const float*)d_in[10];
  const float* ffn_in_w = (const float*)d_in[11];
  const float* ffn_in_b = (const float*)d_in[12];
  const float* cl_w     = (const float*)d_in[13];
  const float* cl_b     = (const float*)d_in[14];
  const float* ffn_out_w= (const float*)d_in[15];
  const float* ffn_out_b= (const float*)d_in[16];

  char* ws = (char*)d_ws;
  float*     w_scale = (float*)ws;                         // 256
  float*     w_shift = w_scale + 256;                      // 256
  _Float16*  wqkv_h  = (_Float16*)(ws + 4096);             // 768*256
  _Float16*  win_h   = wqkv_h + 768 * 256;                 // 256*256
  _Float16*  wout_h  = win_h + 256 * 256;                  // 256*256
  float*     xa      = (float*)(ws + 4096 + 655360);       // 25,690,112 f32
  _Float16*  y1h     = (_Float16*)(ws + 4096 + 655360 + 102760448);
  _Float16*  y2h     = y1h + (size_t)BB * CC * HWIMG;

  bn_stats<<<CC, 256, 0, stream>>>(x, bn_gamma, bn_beta, w_scale, w_shift);
  pack_w<<<1280, 256, 0, stream>>>(qkv_w, ffn_in_w, ffn_out_w,
                                   wqkv_h, win_h, wout_h);
  attn_win<<<NWIN, 256, SMEM_ATTN, stream>>>(
      x, w_scale, w_shift, wqkv_h, qkv_b,
      conv3_w, conv3_b, conv5_w, conv5_b, conv7_w, conv7_b, xa);
  ffn_gemm<float><<<784, 256, 0, stream>>>(xa, win_h, ffn_in_b,
                                           nullptr, nullptr, y1h);
  dw_silu<<<BB * CC * HWIMG / 256, 256, 0, stream>>>(y1h, cl_w, cl_b, y2h);
  ffn_gemm<_Float16><<<784, 256, 0, stream>>>(y2h, wout_h, ffn_out_b,
                                              xa, (float*)d_out, nullptr);
}
